// Model_34686155882827
// MI455X (gfx1250) — matrix operand order repaired and checked by reading, not compiled, not run on hardware
//
#include <hip/hip_runtime.h>

// Problem constants (match reference)
#define N_NODES   50000
#define N_EDGES   800000
#define F_IN      512
#define NHID      256
#define N_CLASSES 10
#define N_GRAPHS  64
#define MPAD      50176   // 196 * 256, padded row count for guard-free WMMA GEMM

typedef __attribute__((ext_vector_type(16))) __bf16 v16bf;
typedef __attribute__((ext_vector_type(8)))  float  v8f;

union Frag16 {
  v16bf    v;
  unsigned u[8];
};

__device__ __forceinline__ unsigned short f32_to_bf16(float f) {
  unsigned u = __float_as_uint(f);
  u += 0x7FFFu + ((u >> 16) & 1u);   // round-to-nearest-even
  return (unsigned short)(u >> 16);
}

// ---------------------------------------------------------------------------
// Degree / norm
// ---------------------------------------------------------------------------
__global__ void k_fill_f32(float* __restrict__ p, float v, int n) {
  int i = blockIdx.x * blockDim.x + threadIdx.x;
  if (i < n) p[i] = v;
}
__global__ void k_deg_edge(const int* __restrict__ dst, float* __restrict__ deg, int e) {
  int i = blockIdx.x * blockDim.x + threadIdx.x;
  if (i < e) atomicAdd(&deg[dst[i]], 1.0f);
}
__global__ void k_rsqrt_inplace(float* __restrict__ d, int n) {
  int i = blockIdx.x * blockDim.x + threadIdx.x;
  if (i < n) d[i] = rsqrtf(d[i]);
}

// ---------------------------------------------------------------------------
// f32 -> bf16 conversions
// ---------------------------------------------------------------------------
__global__ void k_cvt_x_pad(const float* __restrict__ x, unsigned short* __restrict__ xb) {
  int i = blockIdx.x * blockDim.x + threadIdx.x;
  if (i >= MPAD * F_IN) return;
  int r = i >> 9;  // / F_IN (512)
  xb[i] = (r < N_NODES) ? f32_to_bf16(x[i]) : (unsigned short)0;
}
__global__ void k_cvt(const float* __restrict__ in, unsigned short* __restrict__ out, int n) {
  int i = blockIdx.x * blockDim.x + threadIdx.x;
  if (i < n) out[i] = f32_to_bf16(in[i]);
}

// ---------------------------------------------------------------------------
// WMMA bf16 GEMM:  C[MPAD x 256] (f32) = A[MPAD x K] (bf16) @ B[K x 256] (bf16)
// Block: 256 threads = 8 waves. Block tile 256x64.
// Wave tile 32x64: 2 A-frags x 4 B-frags = 8 WMMAs per K-step (B frag reused).
// ---------------------------------------------------------------------------
__global__ __launch_bounds__(256)
void k_gemm_bf16(const unsigned short* __restrict__ A,
                 const unsigned short* __restrict__ B,
                 float* __restrict__ C,
                 int K)
{
  __shared__ unsigned short lA[256][34];   // [row][k], padded stride (17.4 KB)
  __shared__ unsigned short lB[64][34];    // transposed: [n][k], padded stride (4.4 KB)

  const int tid  = threadIdx.x;
  const int wave = tid >> 5;
  const int lane = tid & 31;
  const int m0   = blockIdx.x * 256;
  const int n0   = blockIdx.y * 64;

  v8f acc[2][4];
#pragma unroll
  for (int mt = 0; mt < 2; ++mt)
#pragma unroll
    for (int nt = 0; nt < 4; ++nt)
#pragma unroll
      for (int j = 0; j < 8; ++j) acc[mt][nt][j] = 0.0f;

  // A-tile load mapping: thread -> (row, 16-elem chunk); 2 chunks per thread
  const int ar = tid >> 1;
  const int ac = (tid & 1) * 16;
  // B-tile load mapping: thread -> (k-row, 8-elem chunk)
  const int bk = tid >> 3;
  const int bn = (tid & 7) * 8;

  for (int k0 = 0; k0 < K; k0 += 32) {
    // Stage A: 256x32 bf16 (aligned 32B global loads, 2 rows per thread)
#pragma unroll
    for (int half = 0; half < 2; ++half) {
      const int row = ar + half * 128;
      const uint4* ga = (const uint4*)(A + (size_t)(m0 + row) * K + (k0 + ac));
      uint4 q0 = ga[0];
      uint4 q1 = ga[1];
      unsigned* la = (unsigned*)&lA[row][ac];
      la[0] = q0.x; la[1] = q0.y; la[2] = q0.z; la[3] = q0.w;
      la[4] = q1.x; la[5] = q1.y; la[6] = q1.z; la[7] = q1.w;
    }
    // Stage B transposed: 32x64 bf16 -> lB[n][k]
    {
      uint4 qb = *(const uint4*)(B + (size_t)(k0 + bk) * NHID + (n0 + bn));
      unsigned short e0 = (unsigned short)(qb.x & 0xFFFFu), e1 = (unsigned short)(qb.x >> 16);
      unsigned short e2 = (unsigned short)(qb.y & 0xFFFFu), e3 = (unsigned short)(qb.y >> 16);
      unsigned short e4 = (unsigned short)(qb.z & 0xFFFFu), e5 = (unsigned short)(qb.z >> 16);
      unsigned short e6 = (unsigned short)(qb.w & 0xFFFFu), e7 = (unsigned short)(qb.w >> 16);
      lB[bn + 0][bk] = e0; lB[bn + 1][bk] = e1;
      lB[bn + 2][bk] = e2; lB[bn + 3][bk] = e3;
      lB[bn + 4][bk] = e4; lB[bn + 5][bk] = e5;
      lB[bn + 6][bk] = e6; lB[bn + 7][bk] = e7;
    }
    __syncthreads();

    // Build 2 A fragments (16x32 bf16, ISA VGPR layout):
    //   lane<16: M=lane, K in {0..7, 16..23}; lane>=16: M=lane-16, K in {8..15, 24..31}
    Frag16 af[2];
    {
      const int am = lane & 15;
      const int kh = (lane >> 4) * 8;
#pragma unroll
      for (int mt = 0; mt < 2; ++mt) {
        const unsigned short* arow = &lA[wave * 32 + mt * 16 + am][0];
#pragma unroll
        for (int j = 0; j < 4; ++j) af[mt].u[j]     = *(const unsigned*)(arow + kh + 2 * j);
#pragma unroll
        for (int j = 0; j < 4; ++j) af[mt].u[4 + j] = *(const unsigned*)(arow + 16 + kh + 2 * j);
      }
    }

    // Build 4 B fragments (32x16 bf16: N=lane&15, K in {kb..kb+7, 16+kb..16+kb+7}, kb=(lane>>4)*8); each
    // feeds two WMMAs (reused across the wave's two row-tiles).
    const int nb = lane & 15;
    const int kb = (lane >> 4) * 8;
#pragma unroll
    for (int nt = 0; nt < 4; ++nt) {
      Frag16 bfr;
      const unsigned short* brow = &lB[nt * 16 + nb][kb];
#pragma unroll
      for (int j = 0; j < 8; ++j) bfr.u[j] = *(const unsigned*)(brow + 2 * j + (j >= 4 ? 8 : 0));
      acc[0][nt] = __builtin_amdgcn_wmma_f32_16x16x32_bf16(
          false, af[0].v, false, bfr.v, (short)0, acc[0][nt], false, false);
      acc[1][nt] = __builtin_amdgcn_wmma_f32_16x16x32_bf16(
          false, af[1].v, false, bfr.v, (short)0, acc[1][nt], false, false);
    }
    __syncthreads();
  }

  // Store C: f32 16x16 D layout -> lane<16 rows r, lane>=16 rows 8+r
  const int cn = lane & 15;
#pragma unroll
  for (int mt = 0; mt < 2; ++mt) {
    const int rb = m0 + wave * 32 + mt * 16 + (lane >> 4) * 8;
#pragma unroll
    for (int nt = 0; nt < 4; ++nt) {
      const int col = n0 + nt * 16 + cn;
#pragma unroll
      for (int r = 0; r < 8; ++r)
        C[(size_t)(rb + r) * NHID + col] = acc[mt][nt][r];
    }
  }
}

// ---------------------------------------------------------------------------
// Aggregation: self term, edge scatter-add (L2-resident atomics), bias+ReLU
// ---------------------------------------------------------------------------
__global__ void k_self_init(const float* __restrict__ hw, const float* __restrict__ dinv,
                            float* __restrict__ agg) {
  int i = blockIdx.x * blockDim.x + threadIdx.x;
  if (i >= N_NODES * NHID) return;
  int nd = i >> 8;
  float di = dinv[nd];
  agg[i] = hw[i] * di * di;
}

__global__ __launch_bounds__(256)
void k_edge_agg(const int* __restrict__ esrc, const int* __restrict__ edst,
                const float* __restrict__ dinv,
                const float* __restrict__ hw, float* __restrict__ agg)
{
  int wid = (blockIdx.x * blockDim.x + threadIdx.x) >> 5;   // one wave per edge
  if (wid >= N_EDGES) return;
  int lane = threadIdx.x & 31;
  int s = esrc[wid], d = edst[wid];
  float nrm = dinv[s] * dinv[d];
  const float4* hs = (const float4*)(hw + (size_t)s * NHID);
  float* ad = agg + (size_t)d * NHID + lane * 8;
  float4 a = hs[lane * 2];
  float4 b = hs[lane * 2 + 1];
  atomicAdd(ad + 0, a.x * nrm); atomicAdd(ad + 1, a.y * nrm);
  atomicAdd(ad + 2, a.z * nrm); atomicAdd(ad + 3, a.w * nrm);
  atomicAdd(ad + 4, b.x * nrm); atomicAdd(ad + 5, b.y * nrm);
  atomicAdd(ad + 6, b.z * nrm); atomicAdd(ad + 7, b.w * nrm);
}

__global__ void k_relu_store(const float* __restrict__ agg, const float* __restrict__ bias,
                             unsigned short* __restrict__ h) {
  int i = blockIdx.x * blockDim.x + threadIdx.x;
  if (i >= MPAD * NHID) return;
  int nd = i >> 8;
  int f  = i & 255;
  float v = 0.0f;
  if (nd < N_NODES) v = fmaxf(agg[i] + bias[f], 0.0f);
  h[i] = f32_to_bf16(v);   // pad rows written as 0 so next GEMM is guard-free
}

// ---------------------------------------------------------------------------
// Pooling (uint atomicMax valid for ReLU outputs >= 0) and head
// ---------------------------------------------------------------------------
__global__ void k_zero_u32(unsigned* __restrict__ p, int n) {
  int i = blockIdx.x * blockDim.x + threadIdx.x;
  if (i < n) p[i] = 0u;
}
__global__ void k_pool_max(const unsigned short* __restrict__ h, const int* __restrict__ batch,
                           unsigned* __restrict__ g) {
  int i = blockIdx.x * blockDim.x + threadIdx.x;
  if (i >= N_NODES * NHID) return;
  int nd = i >> 8;
  int f  = i & 255;
  unsigned bits = ((unsigned)h[i]) << 16;  // non-negative f32 bit pattern
  atomicMax(&g[(size_t)batch[nd] * NHID + f], bits);
}

__global__ void k_head(const unsigned* __restrict__ g, const float* __restrict__ Wl,
                       const float* __restrict__ bl, float* __restrict__ out) {
  int gi = threadIdx.x;   // one thread per graph, blockDim = 64
  if (gi >= N_GRAPHS) return;
  float logit[N_CLASSES];
#pragma unroll
  for (int c = 0; c < N_CLASSES; ++c) logit[c] = bl[c];
  for (int k = 0; k < NHID; ++k) {
    float gv = __uint_as_float(g[gi * NHID + k]);
#pragma unroll
    for (int c = 0; c < N_CLASSES; ++c) logit[c] += gv * Wl[k * N_CLASSES + c];
  }
  float mx = logit[0];
#pragma unroll
  for (int c = 1; c < N_CLASSES; ++c) mx = fmaxf(mx, logit[c]);
  float s = 0.0f;
#pragma unroll
  for (int c = 0; c < N_CLASSES; ++c) s += expf(logit[c] - mx);
  float lse = mx + logf(s);
#pragma unroll
  for (int c = 0; c < N_CLASSES; ++c) out[gi * N_CLASSES + c] = logit[c] - lse;
}

// ---------------------------------------------------------------------------
// Launch
// ---------------------------------------------------------------------------
extern "C" void kernel_launch(void* const* d_in, const int* in_sizes, int n_in,
                              void* d_out, int out_size, void* d_ws, size_t ws_size,
                              hipStream_t stream)
{
  const float* x    = (const float*)d_in[0];
  const int*   ei   = (const int*)d_in[1];
  const int*   esrc = ei;
  const int*   edst = ei + N_EDGES;
  const int*   batch = (const int*)d_in[2];
  const float* W1 = (const float*)d_in[3];
  const float* b1 = (const float*)d_in[4];
  const float* W2 = (const float*)d_in[5];
  const float* b2 = (const float*)d_in[6];
  const float* W3 = (const float*)d_in[7];
  const float* b3 = (const float*)d_in[8];
  const float* Wl = (const float*)d_in[9];
  const float* bl = (const float*)d_in[10];
  float* out = (float*)d_out;

  char* ws = (char*)d_ws;
  size_t off = 0;
  auto alloc = [&](size_t bytes) -> void* {
    void* p = ws + off;
    off += (bytes + 255) & ~(size_t)255;
    return p;
  };
  unsigned short* xb  = (unsigned short*)alloc((size_t)MPAD * F_IN * 2);   // 51.4 MB
  unsigned short* hb  = (unsigned short*)alloc((size_t)MPAD * NHID * 2);   // 25.7 MB
  float*          hw  = (float*)alloc((size_t)MPAD * NHID * 4);            // 51.4 MB
  float*          agg = (float*)alloc((size_t)MPAD * NHID * 4);            // 51.4 MB
  float*          dinv = (float*)alloc((size_t)N_NODES * 4);
  unsigned short* w1b = (unsigned short*)alloc((size_t)F_IN * NHID * 2);
  unsigned short* w2b = (unsigned short*)alloc((size_t)NHID * NHID * 2);
  unsigned short* w3b = (unsigned short*)alloc((size_t)NHID * NHID * 2);
  unsigned*       g   = (unsigned*)alloc((size_t)N_GRAPHS * NHID * 4);

  const int T = 256;

  // Degrees with self-loops, then dinv = rsqrt(deg)
  k_fill_f32<<<(N_NODES + T - 1) / T, T, 0, stream>>>(dinv, 1.0f, N_NODES);
  k_deg_edge<<<(N_EDGES + T - 1) / T, T, 0, stream>>>(edst, dinv, N_EDGES);
  k_rsqrt_inplace<<<(N_NODES + T - 1) / T, T, 0, stream>>>(dinv, N_NODES);

  // bf16 conversions (x zero-padded to MPAD rows)
  k_cvt_x_pad<<<(MPAD * F_IN + T - 1) / T, T, 0, stream>>>(x, xb);
  k_cvt<<<(F_IN * NHID + T - 1) / T, T, 0, stream>>>(W1, w1b, F_IN * NHID);
  k_cvt<<<(NHID * NHID + T - 1) / T, T, 0, stream>>>(W2, w2b, NHID * NHID);
  k_cvt<<<(NHID * NHID + T - 1) / T, T, 0, stream>>>(W3, w3b, NHID * NHID);

  dim3 ggrid(MPAD / 256, NHID / 64);
  const int nAgg  = (N_NODES * NHID + T - 1) / T;
  const int nRelu = (MPAD * NHID + T - 1) / T;
  const int nEdge = N_EDGES / 8;   // one wave per edge, 8 waves per block

  // Layer 1
  k_gemm_bf16<<<ggrid, T, 0, stream>>>(xb, w1b, hw, F_IN);
  k_self_init<<<nAgg, T, 0, stream>>>(hw, dinv, agg);
  k_edge_agg<<<nEdge, T, 0, stream>>>(esrc, edst, dinv, hw, agg);
  k_relu_store<<<nRelu, T, 0, stream>>>(agg, b1, hb);

  // Layer 2
  k_gemm_bf16<<<ggrid, T, 0, stream>>>(hb, w2b, hw, NHID);
  k_self_init<<<nAgg, T, 0, stream>>>(hw, dinv, agg);
  k_edge_agg<<<nEdge, T, 0, stream>>>(esrc, edst, dinv, hw, agg);
  k_relu_store<<<nRelu, T, 0, stream>>>(agg, b2, hb);

  // Layer 3
  k_gemm_bf16<<<ggrid, T, 0, stream>>>(hb, w3b, hw, NHID);
  k_self_init<<<nAgg, T, 0, stream>>>(hw, dinv, agg);
  k_edge_agg<<<nEdge, T, 0, stream>>>(esrc, edst, dinv, hw, agg);
  k_relu_store<<<nRelu, T, 0, stream>>>(agg, b3, hb);

  // Global max pool + classifier head
  k_zero_u32<<<(N_GRAPHS * NHID + T - 1) / T, T, 0, stream>>>(g, N_GRAPHS * NHID);
  k_pool_max<<<(N_NODES * NHID + T - 1) / T, T, 0, stream>>>(hb, batch, g);
  k_head<<<1, 64, 0, stream>>>(g, Wl, bl, out);
}
